// TinyRNN_30030411333838
// MI455X (gfx1250) — matrix operand order repaired and checked by reading, not compiled, not run on hardware
//
#include <hip/hip_runtime.h>

// ---------------------------------------------------------------------------
// TinyRNN for MI455X (gfx1250, wave32, WMMA)  --  B=16, S=256, V=32000, H=1024
//  K1: convert Whh/Why fp32 -> bf16 (ws)
//  K2: gather emb (B,S,H) fp32 (ws)
//  K3: init per-launch sync state (h ping-pong buffers + arrival counter)
//  K4: recurrence across 16 WGPs: Whh slice resident in LDS (128KB, padded),
//      per-step device-scope arrival-counter barrier, v_wmma f32<-bf16
//  K5: (4096x1024)x(1024x32000) bf16 WMMA GEMM, 32x256 tile/block -> fp32
// ---------------------------------------------------------------------------

typedef __attribute__((ext_vector_type(16))) __bf16 v16bf;
typedef __attribute__((ext_vector_type(8)))  __bf16 v8bf;
typedef __attribute__((ext_vector_type(8)))  float  v8f;

#define RNN_B 16
#define RNN_S 256
#define RNN_V 32000
#define RNN_H 1024
#define NWG   16        // recurrence workgroups (one WGP each)
#define LDPAD 8         // bf16 elements of LDS row padding (bank-conflict fix)

__device__ __forceinline__ float fw_tanh(float x) {
    const float BP = 8192.0f / 32767.0f;
    float xc = fminf(fmaxf(x, -1.0f), 1.0f);
    float a  = fabsf(xc);
    float pw = BP + (a - BP) * 0.75f;
    return (a < BP) ? xc : copysignf(pw, xc);
}

// 16x32 bf16 A-fragment from a row-major [16 x ld] tile (LDS or global).
// lane (m | sel<<4): elems[0..15]=A[m][k0+16*sel..+16) (contiguous K order, matches B)
__device__ __forceinline__ v16bf load_a_frag(const __bf16* __restrict__ base,
                                             int m, int sel, int k0, int ld) {
    const __bf16* row = base + (size_t)m * ld + k0;
    v8bf lo = *(const v8bf*)(row + sel * 16);
    v8bf hi = *(const v8bf*)(row + sel * 16 + 8);
    v16bf a;
#pragma unroll
    for (int i = 0; i < 8; ++i) { a[i] = lo[i]; a[i + 8] = hi[i]; }
    return a;
}

// 32x16 bf16 B-fragment, column (= row of W) pointer given; lane (n|sel<<4)
// holds K [k0+16*sel, +16). Two 16B loads (LDS rows padded -> only 16B aligned).
__device__ __forceinline__ v16bf load_b_frag(const __bf16* __restrict__ wrow,
                                             int k0, int sel) {
    const __bf16* p = wrow + k0 + sel * 16;
    v8bf lo = *(const v8bf*)(p);
    v8bf hi = *(const v8bf*)(p + 8);
    v16bf b;
#pragma unroll
    for (int i = 0; i < 8; ++i) { b[i] = lo[i]; b[i + 8] = hi[i]; }
    return b;
}

// -------------------------- K1: fp32 -> bf16 --------------------------------
__global__ void cvt_f32_to_bf16(const float* __restrict__ in,
                                __bf16* __restrict__ out, int n) {
    int i = blockIdx.x * blockDim.x + threadIdx.x;
    if (i < n) out[i] = (__bf16)in[i];
}

// -------------------------- K2: embedding gather ----------------------------
__global__ void gather_emb(const int* __restrict__ xseq,
                           const float* __restrict__ Wxh,
                           float* __restrict__ emb) {
    int tid = blockIdx.x * blockDim.x + threadIdx.x;   // exact grid: B*S*H
    int hh  = tid & (RNN_H - 1);
    int r   = tid >> 10;                               // b*S+s
    int idx = xseq[r];
    emb[tid] = Wxh[(size_t)hh * RNN_V + idx];
}

// -------------------------- K3: per-launch sync init ------------------------
__global__ void init_sync(__bf16* __restrict__ hglob, unsigned* __restrict__ ctr) {
    int i = blockIdx.x * blockDim.x + threadIdx.x;     // exact grid: 2*B*H
    hglob[i] = (__bf16)0.0f;
    if (i == 0) *ctr = 0u;
}

// -------------------------- K4: recurrence ----------------------------------
// 16 blocks x 128 threads (4 waves). Block g owns output columns [64g, 64g+64):
// its 64x1024 Whh slice lives in LDS for the whole kernel (zero steady-state
// weight traffic). Wave w computes one 16-col tile; two accumulators break the
// WMMA RAW chain. h ping-pongs between two global bf16 buffers; steps are
// separated by a device-scope atomic arrival barrier (monotonic counter,
// reset each launch by K3).
__global__ __launch_bounds__(128)
void rnn_recurrence(const float* __restrict__ emb,    // (B*S, H) fp32
                    const __bf16* __restrict__ WhhB,  // (H, H) bf16, rows = out col j
                    __bf16* __restrict__ hall,        // (B*S, H) bf16 out
                    float* __restrict__ hfinal,       // (B, H) fp32 out
                    __bf16* __restrict__ hglob,       // 2 x (B, H) bf16 ping-pong
                    unsigned* __restrict__ ctr) {     // arrival counter
    extern __shared__ __attribute__((aligned(32))) __bf16 Bsl[]; // 64 x (H+LDPAD)
    const int LDB  = RNN_H + LDPAD;
    const int tid  = threadIdx.x;
    const int lane = tid & 31;
    const int wave = tid >> 5;          // 0..3
    const int j0g  = blockIdx.x * 64;   // first output column of this block
    const int m    = lane & 15;
    const int sel  = lane >> 4;
    const int nB   = lane & 15;

    // one-time: stage this block's Whh slice into LDS (64 rows x 1024, padded)
    for (int c = tid; c < 64 * 128; c += 128) {
        int row = c >> 7;
        int col = (c & 127) * 8;
        *(v8bf*)&Bsl[row * LDB + col] =
            *(const v8bf*)&WhhB[(size_t)(j0g + row) * RNN_H + col];
    }
    __syncthreads();

    const int jg = j0g + wave * 16 + nB;             // this lane's output column
    const __bf16* wrow = &Bsl[(wave * 16 + nB) * LDB];

    for (int t = 0; t < RNN_S; ++t) {
        const __bf16* hcur = hglob + (size_t)(t & 1) * (RNN_B * RNN_H);
        __bf16*       hnxt = hglob + (size_t)((t + 1) & 1) * (RNN_B * RNN_H);

        v8f accA = {}; v8f accB = {};
#pragma unroll 2
        for (int k0 = 0; k0 < RNN_H; k0 += 64) {
            v16bf A0 = load_a_frag(hcur, m, sel, k0,      RNN_H);
            v16bf A1 = load_a_frag(hcur, m, sel, k0 + 32, RNN_H);
            v16bf B0 = load_b_frag(wrow, k0,      sel);
            v16bf B1 = load_b_frag(wrow, k0 + 32, sel);
            accA = __builtin_amdgcn_wmma_f32_16x16x32_bf16(false, A0, false, B0,
                                                           (short)0, accA, false, false);
            accB = __builtin_amdgcn_wmma_f32_16x16x32_bf16(false, A1, false, B1,
                                                           (short)0, accB, false, false);
        }

        const int baseM = sel * 8;
#pragma unroll
        for (int r8 = 0; r8 < 8; ++r8) {
            int   bb = baseM + r8;               // batch index == C-tile row
            int   rg = bb * RNN_S + t;           // row in emb / hall
            float h  = fw_tanh(accA[r8] + accB[r8] + emb[(size_t)rg * RNN_H + jg]);
            hall[(size_t)rg * RNN_H + jg] = (__bf16)h;
            hnxt[bb * RNN_H + jg]         = (__bf16)h;
            if (t == RNN_S - 1) hfinal[bb * RNN_H + jg] = h;
        }

        if (t != RNN_S - 1) {
            __threadfence();                     // our h_next stores visible @ agent
            __syncthreads();                     // whole block fenced
            if (tid == 0)
                __hip_atomic_fetch_add(ctr, 1u, __ATOMIC_RELEASE,
                                       __HIP_MEMORY_SCOPE_AGENT);
            unsigned tgt = (unsigned)NWG * (unsigned)(t + 1);
            while (__hip_atomic_load(ctr, __ATOMIC_ACQUIRE,
                                     __HIP_MEMORY_SCOPE_AGENT) < tgt)
                __builtin_amdgcn_s_sleep(1);
        }
    }
}

// -------------------------- K5: logits GEMM ---------------------------------
// grid = (125, 128); block = 256 (8 waves). Block tile: 32 rows x 256 cols.
// A (32x1024 bf16, padded) staged in LDS; wave w -> cols [bx*256+32w, +32),
// computing a 2x2 grid of 16x16 tiles (4 accumulators, 4 WMMA / K-iter).
__global__ __launch_bounds__(256)
void gemm_logits(const __bf16* __restrict__ hall,   // (4096, 1024) bf16
                 const __bf16* __restrict__ WhyB,   // (32000, 1024) bf16
                 float* __restrict__ out) {         // (4096, 32000) fp32
    extern __shared__ __attribute__((aligned(32))) __bf16 Atile[]; // 32 x (H+LDPAD)
    const int LDA  = RNN_H + LDPAD;
    const int tid  = threadIdx.x;
    const int lane = tid & 31;
    const int wave = tid >> 5;          // 0..7
    const int r0   = blockIdx.y * 32;   // output row block (2 M-tiles)

    for (int c = tid; c < 32 * 128; c += 256) {
        int row = c >> 7;
        int col = (c & 127) * 8;
        *(v8bf*)&Atile[row * LDA + col] =
            *(const v8bf*)&hall[(size_t)(r0 + row) * RNN_H + col];
    }
    __syncthreads();

    const int m   = lane & 15;
    const int sel = lane >> 4;
    const int nB  = lane & 15;
    const int v0  = blockIdx.x * 256 + wave * 32;

    const __bf16* brow0 = WhyB + (size_t)(v0 + nB) * RNN_H + sel * 16;
    const __bf16* brow1 = WhyB + (size_t)(v0 + 16 + nB) * RNN_H + sel * 16;

    v8f a00 = {}, a01 = {}, a10 = {}, a11 = {};
#pragma unroll 4
    for (int k0 = 0; k0 < RNN_H; k0 += 32) {
        if (k0 + 256 < RNN_H) {          // stream-ahead on Why rows
            __builtin_prefetch((const void*)(brow0 + k0 + 256), 0, 0);
            __builtin_prefetch((const void*)(brow1 + k0 + 256), 0, 0);
        }
        v16bf A0 = load_a_frag(Atile,            m, sel, k0, LDA);
        v16bf A1 = load_a_frag(Atile + 16 * LDA, m, sel, k0, LDA);
        v16bf B0 = *(const v16bf*)(brow0 + k0);
        v16bf B1 = *(const v16bf*)(brow1 + k0);
        a00 = __builtin_amdgcn_wmma_f32_16x16x32_bf16(false, A0, false, B0,
                                                      (short)0, a00, false, false);
        a01 = __builtin_amdgcn_wmma_f32_16x16x32_bf16(false, A0, false, B1,
                                                      (short)0, a01, false, false);
        a10 = __builtin_amdgcn_wmma_f32_16x16x32_bf16(false, A1, false, B0,
                                                      (short)0, a10, false, false);
        a11 = __builtin_amdgcn_wmma_f32_16x16x32_bf16(false, A1, false, B1,
                                                      (short)0, a11, false, false);
    }

    const int baseM = sel * 8;
#pragma unroll
    for (int r8 = 0; r8 < 8; ++r8) {
        size_t row0 = (size_t)(r0 + baseM + r8) * RNN_V;
        size_t row1 = (size_t)(r0 + 16 + baseM + r8) * RNN_V;
        out[row0 + v0 + nB]      = a00[r8];
        out[row0 + v0 + 16 + nB] = a01[r8];
        out[row1 + v0 + nB]      = a10[r8];
        out[row1 + v0 + 16 + nB] = a11[r8];
    }
}

// ---------------------------------------------------------------------------
extern "C" void kernel_launch(void* const* d_in, const int* in_sizes, int n_in,
                              void* d_out, int out_size, void* d_ws, size_t ws_size,
                              hipStream_t stream) {
    const int*   xseq = (const int*)d_in[0];     // (B,S) int32
    const float* Wxh  = (const float*)d_in[1];   // (H,V)
    const float* Whh  = (const float*)d_in[2];   // (H,H)
    const float* Why  = (const float*)d_in[3];   // (V,H)
    float* out = (float*)d_out;                  // logits (B,S,V) ++ h_final (B,H)

    char* ws = (char*)d_ws;
    float*    emb   = (float*)(ws);                                  // 16 MiB
    __bf16*   hall  = (__bf16*)(ws + (size_t)16  * 1048576);         //  8 MiB
    __bf16*   WhhB  = (__bf16*)(ws + (size_t)24  * 1048576);         //  2 MiB
    __bf16*   WhyB  = (__bf16*)(ws + (size_t)32  * 1048576);         // 62.5 MiB
    __bf16*   hglob = (__bf16*)(ws + (size_t)100 * 1048576);         // 64 KiB
    unsigned* ctr   = (unsigned*)(ws + (size_t)101 * 1048576);       //  4 B

    const int nWhh  = RNN_H * RNN_H;          // 1,048,576
    const int nWhy  = RNN_V * RNN_H;          // 32,768,000
    const int nEmb  = RNN_B * RNN_S * RNN_H;  // 4,194,304
    const int nHgl  = 2 * RNN_B * RNN_H;      // 32,768
    float* hfinal = out + (size_t)RNN_B * RNN_S * RNN_V;  // +131,072,000

    const size_t ldsRec  = (size_t)64 * (RNN_H + LDPAD) * sizeof(__bf16); // 132,096 B
    const size_t ldsGemm = (size_t)32 * (RNN_H + LDPAD) * sizeof(__bf16); //  66,048 B

    cvt_f32_to_bf16<<<nWhh / 256, 256, 0, stream>>>(Whh, WhhB, nWhh);
    cvt_f32_to_bf16<<<nWhy / 256, 256, 0, stream>>>(Why, WhyB, nWhy);
    gather_emb<<<nEmb / 256, 256, 0, stream>>>(xseq, Wxh, emb);
    init_sync<<<nHgl / 256, 256, 0, stream>>>(hglob, ctr);
    rnn_recurrence<<<NWG, 128, ldsRec, stream>>>(emb, WhhB, hall, hfinal, hglob, ctr);
    gemm_logits<<<dim3(125, 128), 256, ldsGemm, stream>>>(hall, WhyB, out);
}
